// VQVAE_1597727834319
// MI455X (gfx1250) — matrix operand order repaired and checked by reading, not compiled, not run on hardware
//
#include <hip/hip_runtime.h>
#include <math.h>

// ---------------------------------------------------------------------------
// VQ-VAE forward for MI455X (gfx1250).
// Conv layers -> implicit GEMM on v_wmma_f32_16x16x32_bf16 (f32 accumulate),
// double-buffered LDS pipeline (global gathers for tile k+1 issued while tile
// k's WMMAs run). LDS tiles stored in WMMA fragment order so fragment loads
// are contiguous ds_load_b128 pairs. TDM (tensor_load_to_lds) used to warm
// GL2 with the large weight tensors before the GEMM chain.
// ---------------------------------------------------------------------------

typedef __attribute__((ext_vector_type(16))) __bf16 v16bf;
typedef __attribute__((ext_vector_type(8)))  float  v8f;

#define BNEPS 1e-5f

static __device__ __forceinline__ unsigned pack_bf16(float a, float b) {
  unsigned short lo = __builtin_bit_cast(unsigned short, (__bf16)a);
  unsigned short hi = __builtin_bit_cast(unsigned short, (__bf16)b);
  return (unsigned)lo | ((unsigned)hi << 16);
}

// GEMM modes
//  0: forward conv k4 s2 (VALID)       K = I*4
//  1: 1x1 conv                         K = I
//  2: tconv (k4,s2,p1) even outputs    K = I*2   (taps w[3], w[1])
//  3: tconv (k4,s2,p1) odd  outputs    K = I*2   (taps w[2], w[0])
//
// Block tile 64(M) x 128(N); 8 waves as 4(M) x 2(N); each wave: one A frag
// reused across 4 WMMAs (16x64 output slab), K-step 32, 2-deep LDS pipeline.
template <int MODE>
__global__ __launch_bounds__(256) void conv_gemm_kernel(
    const float* __restrict__ X, const float* __restrict__ W,
    const float* __restrict__ bias, const float* __restrict__ cond,
    float* __restrict__ Y,
    int O, int I, int Ktot, int Lin, int Lout, int Ldec) {
  const int Ntot  = 8 * Ldec;
  const int oBase = blockIdx.y * 64;
  const int nBase = blockIdx.x * 128;
  const int tid   = threadIdx.x;
  const int wave  = tid >> 5, lane = tid & 31;
  const int wm    = wave >> 1, wn = wave & 1;
  const int lm    = lane & 15, lh = lane >> 4;

  // batch boundary inside this block's N range (Ldec >= 1022 > 128: at most 1)
  const int bTop   = nBase / Ldec;
  const int bBound = (bTop + 1) * Ldec;

  // Double-buffered fragment-ordered tiles: [buf][subtile][lane][8 dwords]
  __shared__ __attribute__((aligned(32))) unsigned AsU[2][4][32][8];
  __shared__ __attribute__((aligned(32))) unsigned BsU[2][8][32][8];

  v8f acc[4] = {};
  float a0[4], a1[4];   // staged weight pairs
  float x0[8], x1[8];   // staged activation pairs

  auto gatherA = [&](int kb) {
#pragma unroll
    for (int r = 0; r < 4; ++r) {
      int pi = tid + r * 256;
      int m = pi >> 4, kk = (pi & 15) * 2;
      int o = oBase + m, Kg = kb + kk;
      if (MODE <= 1) {
        const float2 w2 = *(const float2*)(W + (size_t)o * Ktot + Kg);
        a0[r] = w2.x; a1[r] = w2.y;
      } else {
        int i = Kg >> 1;                       // Kg even: pair covers j=0,1
        const float4 w4 = *(const float4*)(W + ((size_t)o * I + i) * 4);
        if (MODE == 2) { a0[r] = w4.w; a1[r] = w4.y; }   // w[3], w[1]
        else           { a0[r] = w4.z; a1[r] = w4.x; }   // w[2], w[0]
      }
    }
  };
  auto gatherB = [&](int kb) {
#pragma unroll
    for (int r = 0; r < 8; ++r) {
      int pi = tid + r * 256;
      int nl = pi & 127, kk = (pi >> 7) * 2;
      int Kg = kb + kk, n = nBase + nl;
      float f0 = 0.f, f1 = 0.f;
      if (n < Ntot) {
        int b = (n >= bBound) ? bTop + 1 : bTop;
        int t = n - b * Ldec;
        if (MODE == 0) {
          int ci = Kg >> 2, k2 = Kg & 3;       // k2 in {0,2}; pair contiguous
          const float* p0 = X + ((size_t)b * I + ci) * Lin + 2 * t + k2;
          f0 = p0[0]; f1 = p0[1];
        } else if (MODE == 1) {
          const float* p0 = X + ((size_t)b * I + Kg) * Lin + t;
          f0 = p0[0]; f1 = p0[(size_t)Lin];
        } else if (MODE == 2) {
          int i = Kg >> 1;
          const float* p0 = X + ((size_t)b * I + i) * Lin + t;
          f0 = (t >= 1) ? p0[-1] : 0.f;
          f1 = p0[0];
        } else {
          int i = Kg >> 1;
          const float* p0 = X + ((size_t)b * I + i) * Lin + t;
          f0 = p0[0];
          f1 = (t + 1 < Lin) ? p0[1] : 0.f;
        }
      }
      x0[r] = f0; x1[r] = f1;
    }
  };
  auto commit = [&](int buf) {
#pragma unroll
    for (int r = 0; r < 4; ++r) {
      int pi = tid + r * 256;
      int m = pi >> 4, kk = (pi & 15) * 2;
      int laneA = ((kk >> 3) & 1) * 16 + (m & 15);
      int uidx  = 4 * (kk >> 4) + ((kk & 7) >> 1);
      AsU[buf][m >> 4][laneA][uidx] = pack_bf16(a0[r], a1[r]);
    }
#pragma unroll
    for (int r = 0; r < 8; ++r) {
      int pi = tid + r * 256;
      int nl = pi & 127, kk = (pi >> 7) * 2;
      BsU[buf][nl >> 4][((kk >> 3) & 1) * 16 + (nl & 15)][4 * (kk >> 4) + ((kk & 7) >> 1)] =
          pack_bf16(x0[r], x1[r]);
    }
  };

  // prologue: stage tile 0
  gatherA(0);
  gatherB(0);
  commit(0);
  __syncthreads();

  int cur = 0;
  int kb = 0;
  for (; kb + 32 < Ktot; kb += 32) {
    // fragments for this tile (contiguous 32B per lane -> ds_load_b128 x2)
    const v16bf af  = *(const v16bf*)(&AsU[cur][wm][lane][0]);
    const v16bf bq0 = *(const v16bf*)(&BsU[cur][wn * 4 + 0][lane][0]);
    const v16bf bq1 = *(const v16bf*)(&BsU[cur][wn * 4 + 1][lane][0]);
    const v16bf bq2 = *(const v16bf*)(&BsU[cur][wn * 4 + 2][lane][0]);
    const v16bf bq3 = *(const v16bf*)(&BsU[cur][wn * 4 + 3][lane][0]);
    // issue global gathers for next tile (in flight during WMMAs)
    gatherA(kb + 32);
    gatherB(kb + 32);
    // 4 back-to-back WMMAs, A fragment reused
    acc[0] = __builtin_amdgcn_wmma_f32_16x16x32_bf16(false, af, false, bq0,
                                                     (short)0, acc[0], false, false);
    acc[1] = __builtin_amdgcn_wmma_f32_16x16x32_bf16(false, af, false, bq1,
                                                     (short)0, acc[1], false, false);
    acc[2] = __builtin_amdgcn_wmma_f32_16x16x32_bf16(false, af, false, bq2,
                                                     (short)0, acc[2], false, false);
    acc[3] = __builtin_amdgcn_wmma_f32_16x16x32_bf16(false, af, false, bq3,
                                                     (short)0, acc[3], false, false);
    // pack + store next tile into the other buffer
    commit(cur ^ 1);
    __syncthreads();
    cur ^= 1;
  }
  {  // peeled last tile
    const v16bf af  = *(const v16bf*)(&AsU[cur][wm][lane][0]);
    const v16bf bq0 = *(const v16bf*)(&BsU[cur][wn * 4 + 0][lane][0]);
    const v16bf bq1 = *(const v16bf*)(&BsU[cur][wn * 4 + 1][lane][0]);
    const v16bf bq2 = *(const v16bf*)(&BsU[cur][wn * 4 + 2][lane][0]);
    const v16bf bq3 = *(const v16bf*)(&BsU[cur][wn * 4 + 3][lane][0]);
    acc[0] = __builtin_amdgcn_wmma_f32_16x16x32_bf16(false, af, false, bq0,
                                                     (short)0, acc[0], false, false);
    acc[1] = __builtin_amdgcn_wmma_f32_16x16x32_bf16(false, af, false, bq1,
                                                     (short)0, acc[1], false, false);
    acc[2] = __builtin_amdgcn_wmma_f32_16x16x32_bf16(false, af, false, bq2,
                                                     (short)0, acc[2], false, false);
    acc[3] = __builtin_amdgcn_wmma_f32_16x16x32_bf16(false, af, false, bq3,
                                                     (short)0, acc[3], false, false);
  }

  // ---------------- epilogue: bias (+cond), scatter to [B,O,Lout] ---------
#pragma unroll
  for (int q = 0; q < 4; ++q) {
    int n = nBase + wn * 64 + q * 16 + lm;
    if (n < Ntot) {
      int b = (n >= bBound) ? bTop + 1 : bTop;
      int t = n - b * Ldec;
      size_t pos;
      if (MODE == 2)      pos = 2 * (size_t)t;
      else if (MODE == 3) pos = 2 * (size_t)t + 1;
      else                pos = (size_t)t;
      const size_t base = ((size_t)b * O) * Lout + pos;
#pragma unroll
      for (int r = 0; r < 8; ++r) {
        int o = oBase + wm * 16 + lh * 8 + r;
        float v = acc[q][r] + bias[o];
        if (cond) v += cond[(size_t)b * O + o];
        Y[base + (size_t)o * Lout] = v;
      }
    }
  }
}

// ---------------------------------------------------------------------------
// TDM warm: tensor_load_to_lds of the first 16KB of a weight tensor (1-row 2D
// tile, data_size=4B) -> pulls weights through GL2 ahead of the GEMM chain.
// D# layout per CDNA5 ISA 8.3-8.5.  Builtin arity differs across toolchains.
// ---------------------------------------------------------------------------
typedef __attribute__((ext_vector_type(4))) unsigned tdm_u4;
typedef __attribute__((ext_vector_type(8))) int      tdm_i8;
typedef __attribute__((ext_vector_type(4))) int      tdm_i4;

__global__ __launch_bounds__(32) void tdm_warm_kernel(
    const float* __restrict__ src, float* __restrict__ sink) {
  __shared__ __attribute__((aligned(128))) float tile[4096];
  unsigned long long ga = (unsigned long long)src;
  tdm_u4 g0;
  g0[0] = 1u;                                             // count=1 (user D#)
  g0[1] = 0u;                                             // lds_addr = 0
  g0[2] = (unsigned)(ga & 0xffffffffull);                 // global_addr lo
  g0[3] = (unsigned)((ga >> 32) & 0x01ffffffull) | 0x80000000u;  // hi | type=2
  tdm_i8 g1;
  g1[0] = 0x00020000;           // data_size=4B; no multicast/pad/iterate
  g1[1] = (int)(4096u << 16);   // tensor_dim0[15:0]=4096 (bits 63:48)
  g1[2] = (int)(1u << 16);      // tensor_dim0 hi=0 ; tensor_dim1=1
  g1[3] = (int)(4096u << 16);   // tensor_dim1 hi=0 ; tile_dim0=4096
  g1[4] = 1;                    // tile_dim1=1 ; tile_dim2=0
  g1[5] = 4096;                 // tensor_dim0_stride
  g1[6] = 0;
  g1[7] = 0;
  tdm_i4 gz;
  gz[0] = 0; gz[1] = 0; gz[2] = 0; gz[3] = 0;
#if __clang_major__ >= 23
  tdm_i8 gz8;
  gz8[0] = 0; gz8[1] = 0; gz8[2] = 0; gz8[3] = 0;
  gz8[4] = 0; gz8[5] = 0; gz8[6] = 0; gz8[7] = 0;
  __builtin_amdgcn_tensor_load_to_lds(g0, g1, gz, gz, gz8, 0);
#else
  __builtin_amdgcn_tensor_load_to_lds(g0, g1, gz, gz, 0);
#endif
  __builtin_amdgcn_s_wait_tensorcnt(0);
  if (tile[threadIdx.x] > 1e30f) sink[0] = 1.0f;  // keep tile live; never true
}

// per-channel mean / inv-std over (B=8, L)
__global__ __launch_bounds__(256) void bn_stats_kernel(
    const float* __restrict__ Y, float* __restrict__ mean,
    float* __restrict__ istd, int O, int L) {
  int o = blockIdx.x;
  float s = 0.f, s2 = 0.f;
  for (int b = 0; b < 8; ++b) {
    const float* row = Y + ((size_t)b * O + o) * L;
    for (int l = threadIdx.x; l < L; l += 256) {
      float v = row[l];
      s += v;
      s2 += v * v;
    }
  }
  __shared__ float sh1[256], sh2[256];
  sh1[threadIdx.x] = s;
  sh2[threadIdx.x] = s2;
  __syncthreads();
  for (int st = 128; st > 0; st >>= 1) {
    if (threadIdx.x < st) {
      sh1[threadIdx.x] += sh1[threadIdx.x + st];
      sh2[threadIdx.x] += sh2[threadIdx.x + st];
    }
    __syncthreads();
  }
  if (threadIdx.x == 0) {
    float tot = 8.f * (float)L;
    float m = sh1[0] / tot;
    float var = sh2[0] / tot - m * m;
    mean[o] = m;
    istd[o] = rsqrtf(var + BNEPS);
  }
}

// h = tanh(bn(c)) * sigmoid(bn(g)) in-place into C.  grid: (ceil(L/256), 8*O)
__global__ __launch_bounds__(256) void gated_act_kernel(
    float* __restrict__ C, const float* __restrict__ G,
    const float* __restrict__ mc, const float* __restrict__ ic,
    const float* __restrict__ gc, const float* __restrict__ bc,
    const float* __restrict__ mg, const float* __restrict__ ig,
    const float* __restrict__ gg, const float* __restrict__ bg,
    int O, int L) {
  int y = blockIdx.y;          // b*O + o
  int o = y % O;
  int l = blockIdx.x * 256 + threadIdx.x;
  if (l >= L) return;
  size_t n = (size_t)y * L + l;
  float cv = (C[n] - mc[o]) * ic[o] * gc[o] + bc[o];
  float gv = (G[n] - mg[o]) * ig[o] * gg[o] + bg[o];
  C[n] = tanhf(cv) * (1.f / (1.f + expf(-gv)));
}

// y = bn(y) (optional relu) in place.  grid: (ceil(L/256), 8*O)
__global__ __launch_bounds__(256) void bn_act_kernel(
    float* __restrict__ Y, const float* __restrict__ m,
    const float* __restrict__ is, const float* __restrict__ g,
    const float* __restrict__ b, int O, int L, int relu) {
  int y = blockIdx.y;
  int o = y % O;
  int l = blockIdx.x * 256 + threadIdx.x;
  if (l >= L) return;
  size_t n = (size_t)y * L + l;
  float v = (Y[n] - m[o]) * is[o] * g[o] + b[o];
  if (relu) v = fmaxf(v, 0.f);
  Y[n] = v;
}

// vector quantization: one wave per (b,t); q_st == q numerically in forward
__global__ __launch_bounds__(256) void vq_kernel(
    const float* __restrict__ Z, const float* __restrict__ E,
    float* __restrict__ Qdec, float* __restrict__ Qst,
    float* __restrict__ Qcb, int Tz) {
  __shared__ __attribute__((aligned(16))) float zsh[8][64];
  int wave = threadIdx.x >> 5, lane = threadIdx.x & 31;
  int w = blockIdx.x * 8 + wave;
  int total = 8 * Tz;
  if (w < total) {
    int b = w / Tz, t = w - b * Tz;
    for (int d = lane; d < 64; d += 32)
      zsh[wave][d] = Z[((size_t)b * 64 + d) * Tz + t];
  }
  __syncthreads();
  if (w >= total) return;
  int b = w / Tz, t = w - b * Tz;
  float best = 3.4e38f;
  int bidx = 0;
  const float4* z4 = (const float4*)zsh[wave];
  for (int k = lane; k < 512; k += 32) {
    const float4* e4 = (const float4*)(E + (size_t)k * 64);
    float d2 = 0.f;
#pragma unroll
    for (int d = 0; d < 16; ++d) {
      float4 ev = e4[d], zv = z4[d];
      float dx = zv.x - ev.x, dy = zv.y - ev.y;
      float dz = zv.z - ev.z, dw = zv.w - ev.w;
      d2 = fmaf(dx, dx, fmaf(dy, dy, fmaf(dz, dz, fmaf(dw, dw, d2))));
    }
    if (d2 < best) { best = d2; bidx = k; }
  }
  for (int off = 16; off > 0; off >>= 1) {
    float ob = __shfl_xor(best, off, 32);
    int oi = __shfl_xor(bidx, off, 32);
    if (ob < best || (ob == best && oi < bidx)) { best = ob; bidx = oi; }
  }
  bidx = __shfl(bidx, 0, 32);
  for (int d = lane; d < 64; d += 32) {
    float qv = E[(size_t)bidx * 64 + d];
    size_t off = ((size_t)b * 64 + d) * Tz + t;
    Qdec[off] = qv;
    Qst[off] = qv;
    Qcb[off] = qv;
  }
}

// cond[b,o] = spk_emb[speaker[b]] . condw[o] + condb[o]
__global__ __launch_bounds__(256) void cond_kernel(
    const int* __restrict__ speaker, const float* __restrict__ spk,
    const float* __restrict__ cw, const float* __restrict__ cb,
    float* __restrict__ cond, int O) {
  int i = blockIdx.x * blockDim.x + threadIdx.x;
  if (i >= 8 * O) return;
  int b = i / O, o = i - b * O;
  const float4* s4 = (const float4*)(spk + (size_t)speaker[b] * 64);
  const float4* w4 = (const float4*)(cw + (size_t)o * 64);
  float a = cb[o];
#pragma unroll
  for (int d = 0; d < 16; ++d) {
    float4 sv = s4[d], wv = w4[d];
    a = fmaf(sv.x, wv.x, fmaf(sv.y, wv.y, fmaf(sv.z, wv.z, fmaf(sv.w, wv.w, a))));
  }
  cond[i] = a;
}

// ---------------------------------------------------------------------------

static inline void launch_gemm(int mode, const float* X, const float* W,
                               const float* bias, const float* cond, float* Y,
                               int O, int I, int Ktot, int Lin, int Lout,
                               int Ldec, hipStream_t s) {
  dim3 blk(256);
  dim3 grd((unsigned)((8 * Ldec + 127) / 128), (unsigned)(O / 64));
  switch (mode) {
    case 0: conv_gemm_kernel<0><<<grd, blk, 0, s>>>(X, W, bias, cond, Y, O, I, Ktot, Lin, Lout, Ldec); break;
    case 1: conv_gemm_kernel<1><<<grd, blk, 0, s>>>(X, W, bias, cond, Y, O, I, Ktot, Lin, Lout, Ldec); break;
    case 2: conv_gemm_kernel<2><<<grd, blk, 0, s>>>(X, W, bias, cond, Y, O, I, Ktot, Lin, Lout, Ldec); break;
    default: conv_gemm_kernel<3><<<grd, blk, 0, s>>>(X, W, bias, cond, Y, O, I, Ktot, Lin, Lout, Ldec); break;
  }
}

static inline void run_gated(float* C, float* G, const float* gc,
                             const float* bc, const float* gg, const float* bg,
                             float* meanC, float* istdC, float* meanG,
                             float* istdG, int O, int L, hipStream_t s) {
  bn_stats_kernel<<<O, 256, 0, s>>>(C, meanC, istdC, O, L);
  bn_stats_kernel<<<O, 256, 0, s>>>(G, meanG, istdG, O, L);
  dim3 g((unsigned)((L + 255) / 256), (unsigned)(8 * O));
  gated_act_kernel<<<g, 256, 0, s>>>(C, G, meanC, istdC, gc, bc, meanG, istdG,
                                     gg, bg, O, L);
}

extern "C" void kernel_launch(void* const* d_in, const int* in_sizes, int n_in,
                              void* d_out, int out_size, void* d_ws,
                              size_t ws_size, hipStream_t stream) {
  (void)in_sizes; (void)n_in; (void)out_size; (void)ws_size;
  const float* X  = (const float*)d_in[0];
  const int* SPK  = (const int*)d_in[1];
#define F(i) ((const float*)d_in[i])

  // ---- workspace carve (floats): 3 ping-pong buffers + stats + cond ----
  const size_t BUF = 50331648;  // >= 8*768*8176
  float* ws = (float*)d_ws;
  float* b0 = ws;
  float* b1 = ws + BUF;
  float* b2 = ws + 2 * BUF;
  float* st = ws + 3 * BUF;
  float* meanC = st, * istdC = st + 1024, * meanG = st + 2048, * istdG = st + 3072;
  float* condb = st + 4096;       // 8*768 floats
  float* sink  = st + 12288;      // TDM warm sink (unused value)

  float* out        = (float*)d_out;
  float* out_logits = out;                           // [8,256,8176]
  float* out_qst    = out + (size_t)8 * 256 * 8176;  // [8,64,1022]
  float* out_q      = out_qst + (size_t)8 * 64 * 1022;

  // ---- TDM warm of the large weight tensors (pull into GL2) ----
  tdm_warm_kernel<<<1, 32, 0, stream>>>(F(10), sink);
  tdm_warm_kernel<<<1, 32, 0, stream>>>(F(14), sink);
  tdm_warm_kernel<<<1, 32, 0, stream>>>(F(42), sink);
  tdm_warm_kernel<<<1, 32, 0, stream>>>(F(46), sink);
  tdm_warm_kernel<<<1, 32, 0, stream>>>(F(52), sink);
  tdm_warm_kernel<<<1, 32, 0, stream>>>(F(56), sink);

  // ---- encoder ----
  launch_gemm(0, X, F(2), F(3), nullptr, b0, 768, 80, 320, 8192, 4095, 4095, stream);
  launch_gemm(0, X, F(6), F(7), nullptr, b2, 768, 80, 320, 8192, 4095, 4095, stream);
  run_gated(b0, b2, F(4), F(5), F(8), F(9), meanC, istdC, meanG, istdG, 768, 4095, stream);
  launch_gemm(0, b0, F(10), F(11), nullptr, b1, 768, 768, 3072, 4095, 2046, 2046, stream);
  launch_gemm(0, b0, F(14), F(15), nullptr, b2, 768, 768, 3072, 4095, 2046, 2046, stream);
  run_gated(b1, b2, F(12), F(13), F(16), F(17), meanC, istdC, meanG, istdG, 768, 2046, stream);
  launch_gemm(0, b1, F(18), F(19), nullptr, b0, 768, 768, 3072, 2046, 1022, 1022, stream);
  launch_gemm(0, b1, F(22), F(23), nullptr, b2, 768, 768, 3072, 2046, 1022, 1022, stream);
  run_gated(b0, b2, F(20), F(21), F(24), F(25), meanC, istdC, meanG, istdG, 768, 1022, stream);

  // ---- latent 1x1 + BN -> z in b1 ----
  launch_gemm(1, b0, F(26), F(27), nullptr, b1, 64, 768, 768, 1022, 1022, 1022, stream);
  bn_stats_kernel<<<64, 256, 0, stream>>>(b1, meanC, istdC, 64, 1022);
  {
    dim3 g((1022 + 255) / 256, 8 * 64);
    bn_act_kernel<<<g, 256, 0, stream>>>(b1, meanC, istdC, F(28), F(29), 64, 1022, 0);
  }

  // ---- vector quantization ----
  vq_kernel<<<1022, 256, 0, stream>>>(b1, F(30), b2, out_qst, out_q, 1022);

  // ---- decoder ----
  cond_kernel<<<(8 * 768 + 255) / 256, 256, 0, stream>>>(SPK, F(31), F(40), F(41), condb, 768);
  launch_gemm(2, b2, F(32), F(33), condb, b0, 768, 64, 128, 1022, 2044, 1022, stream);
  launch_gemm(3, b2, F(32), F(33), condb, b0, 768, 64, 128, 1022, 2044, 1022, stream);
  launch_gemm(2, b2, F(36), F(37), condb, b1, 768, 64, 128, 1022, 2044, 1022, stream);
  launch_gemm(3, b2, F(36), F(37), condb, b1, 768, 64, 128, 1022, 2044, 1022, stream);
  run_gated(b0, b1, F(34), F(35), F(38), F(39), meanC, istdC, meanG, istdG, 768, 2044, stream);

  cond_kernel<<<(8 * 768 + 255) / 256, 256, 0, stream>>>(SPK, F(31), F(50), F(51), condb, 768);
  launch_gemm(2, b0, F(42), F(43), condb, b1, 768, 768, 1536, 2044, 4088, 2044, stream);
  launch_gemm(3, b0, F(42), F(43), condb, b1, 768, 768, 1536, 2044, 4088, 2044, stream);
  launch_gemm(2, b0, F(46), F(47), condb, b2, 768, 768, 1536, 2044, 4088, 2044, stream);
  launch_gemm(3, b0, F(46), F(47), condb, b2, 768, 768, 1536, 2044, 4088, 2044, stream);
  run_gated(b1, b2, F(44), F(45), F(48), F(49), meanC, istdC, meanG, istdG, 768, 4088, stream);

  cond_kernel<<<(8 * 768 + 255) / 256, 256, 0, stream>>>(SPK, F(31), F(60), F(61), condb, 768);
  launch_gemm(2, b1, F(52), F(53), condb, b0, 768, 768, 1536, 4088, 8176, 4088, stream);
  launch_gemm(3, b1, F(52), F(53), condb, b0, 768, 768, 1536, 4088, 8176, 4088, stream);
  launch_gemm(2, b1, F(56), F(57), condb, b2, 768, 768, 1536, 4088, 8176, 4088, stream);
  launch_gemm(3, b1, F(56), F(57), condb, b2, 768, 768, 1536, 4088, 8176, 4088, stream);
  run_gated(b0, b2, F(54), F(55), F(58), F(59), meanC, istdC, meanG, istdG, 768, 8176, stream);

  // ---- pre 1x1 + BN + relu ----
  launch_gemm(1, b0, F(62), F(63), nullptr, b1, 256, 768, 768, 8176, 8176, 8176, stream);
  bn_stats_kernel<<<256, 256, 0, stream>>>(b1, meanC, istdC, 256, 8176);
  {
    dim3 g((8176 + 255) / 256, 8 * 256);
    bn_act_kernel<<<g, 256, 0, stream>>>(b1, meanC, istdC, F(64), F(65), 256, 8176, 1);
  }

  // ---- logits 1x1 straight to d_out ----
  launch_gemm(1, b1, F(66), F(67), nullptr, out_logits, 256, 256, 256, 8176, 8176, 8176, stream);
#undef F
}
